// FairscaleMoEBlock_83597243449394
// MI455X (gfx1250) — matrix operand order repaired and checked by reading, not compiled, not run on hardware
//
#include <hip/hip_runtime.h>
#include <math.h>

// ---- problem constants (match reference) ----
#define NTOK 8192      // B*S
#define DDIM 1024      // D
#define EEXP 8         // E
#define HDIM 4096      // H

// ---- tiling ----
#define TM       32            // tokens per block tile
#define HCHUNK   256           // H processed per outer iteration (32 per wave)
#define XS_STRIDE (DDIM + 8)   // LDS stride (bf16 elems), keeps 16B alignment
#define HS_STRIDE (HCHUNK + 8)
#define CSTRIDE  80            // staged-weight col stride in bytes (64B data + 16B pad)

typedef __attribute__((ext_vector_type(16))) __bf16 v16bf;
typedef __attribute__((ext_vector_type(8)))  __bf16 v8bf;
typedef __attribute__((ext_vector_type(8)))  float  v8f;

__device__ __forceinline__ v8f wmma_bf16(v16bf a, v16bf b, v8f c) {
    return __builtin_amdgcn_wmma_f32_16x16x32_bf16(false, a, false, b,
                                                   (short)0, c, false, false);
}

// A-fragment (16x32 bf16) from LDS, per ISA 7.12.2
__device__ __forceinline__ v16bf lds_frag_a(const __bf16* tile, int stride,
                                            int row0, int k, int lane) {
    int r  = row0 + (lane & 15);
    int kk = k + ((lane >> 4) << 3);
    const __bf16* p = tile + r * stride + kk;
    v8bf lo = *(const v8bf*)(p);
    v8bf hi = *(const v8bf*)(p + 16);
    return __builtin_shufflevector(lo, hi, 0,1,2,3,4,5,6,7,8,9,10,11,12,13,14,15);
}

// B-fragment (32x16) from a staged LDS buffer: [col][32 K contiguous], stride 80B
__device__ __forceinline__ v16bf lds_frag_bs(const char* buf, int col, int half) {
    const __bf16* p = (const __bf16*)(buf + col * CSTRIDE + half * 16);
    v8bf lo = *(const v8bf*)(p);
    v8bf hi = *(const v8bf*)(p + 16);
    return __builtin_shufflevector(lo, hi, 0,1,2,3,4,5,6,7,8,9,10,11,12,13,14,15);
}

__device__ __forceinline__ float gelu_exact(float v) {
    return 0.5f * v * (1.0f + erff(v * 0.70710678118654752f));
}

// ---- CDNA5 async global->LDS machinery (ASYNCcnt) ----
// INST_OFFSET is added to BOTH the LDS and the global address (ISA ch.8), and
// our 16B segment stride matches on both sides, so one address pair feeds
// four copies: a 64B burst with a single address setup.
__device__ __forceinline__ void async_copy_64B(unsigned lds_addr,
                                               const void* gptr) {
    unsigned long long ga = (unsigned long long)(uintptr_t)gptr;
    asm volatile(
        "global_load_async_to_lds_b128 %0, %1, off\n\t"
        "global_load_async_to_lds_b128 %0, %1, off offset:16\n\t"
        "global_load_async_to_lds_b128 %0, %1, off offset:32\n\t"
        "global_load_async_to_lds_b128 %0, %1, off offset:48"
        :: "v"(lds_addr), "v"(ga) : "memory");
}
__device__ __forceinline__ void wait_async0()  { asm volatile("s_wait_asynccnt 0x0"  ::: "memory"); }
__device__ __forceinline__ void wait_async4()  { asm volatile("s_wait_asynccnt 0x4"  ::: "memory"); }
__device__ __forceinline__ void wait_async16() { asm volatile("s_wait_asynccnt 0x10" ::: "memory"); }

__device__ __forceinline__ unsigned lds_addr_of(const void* p) {
    return (unsigned)(uintptr_t)p;
}

// stage 32 cols x 64B of W1 rows (wave-private): ONE address setup per lane
__device__ __forceinline__ void stage_w1(unsigned base, const __bf16* w1c, int lane) {
    async_copy_64B(base + (unsigned)(lane * CSTRIDE), w1c + (long)lane * DDIM);
}
// stage 128 cols x 64B of W2 rows (wave-private): four setups per lane
__device__ __forceinline__ void stage_w2(unsigned base, const __bf16* w2c, int lane) {
#pragma unroll
    for (int j = 0; j < 4; ++j) {
        int col = lane + 32 * j;
        async_copy_64B(base + (unsigned)(col * CSTRIDE), w2c + (long)col * HDIM);
    }
}

// ---------------- kernel 1: zero out + counters ----------------
__global__ void moe_init(float* __restrict__ out, int* __restrict__ counts) {
    long i = (long)blockIdx.x * blockDim.x + threadIdx.x;
    long stride = (long)gridDim.x * blockDim.x;
    for (long j = i; j < (long)NTOK * DDIM; j += stride) out[j] = 0.0f;
    if (i < EEXP) counts[i] = 0;
}

// ---------------- kernel 2: f32 -> bf16 conversion ----------------
__global__ void moe_cvt_bf16(const float* __restrict__ src,
                             __bf16* __restrict__ dst, long n) {
    long i = (long)blockIdx.x * blockDim.x + threadIdx.x;
    long stride = (long)gridDim.x * blockDim.x;
    for (; i < n; i += stride) dst[i] = (__bf16)src[i];
}

// ---------------- kernel 3: routing + dispatch compaction ----------------
__global__ void moe_route(const float* __restrict__ x,
                          const float* __restrict__ gw,
                          const float* __restrict__ gu,
                          int*   __restrict__ counts,
                          int*   __restrict__ tok_list,
                          float* __restrict__ wt_list) {
    int lane = threadIdx.x & 31;
    int wave = threadIdx.x >> 5;
    int t = blockIdx.x * 8 + wave;
    if (t >= NTOK) return;

    float acc[EEXP];
#pragma unroll
    for (int e = 0; e < EEXP; ++e) acc[e] = 0.0f;

    const float* xr = x + (long)t * DDIM;
    for (int j = lane; j < DDIM; j += 32) {
        float xv = xr[j];
#pragma unroll
        for (int e = 0; e < EEXP; ++e) acc[e] += xv * gw[e * DDIM + j];
    }
#pragma unroll
    for (int off = 16; off > 0; off >>= 1) {
#pragma unroll
        for (int e = 0; e < EEXP; ++e)
            acc[e] += __shfl_xor(acc[e], off, 32);
    }

    if (lane == 0) {
        float mx = acc[0]; int i1 = 0;
#pragma unroll
        for (int e = 1; e < EEXP; ++e) if (acc[e] > mx) { mx = acc[e]; i1 = e; }
        float g[EEXP]; float se = 0.0f;
#pragma unroll
        for (int e = 0; e < EEXP; ++e) { g[e] = __expf(acc[e] - mx); se += g[e]; }
        float inv = 1.0f / se;
#pragma unroll
        for (int e = 0; e < EEXP; ++e) g[e] *= inv;

        float best2 = -__builtin_inff(); int i2 = 0;
#pragma unroll
        for (int e = 0; e < EEXP; ++e) {
            float u   = fmaxf(gu[(long)t * EEXP + e], 1e-20f);
            float gum = -__logf(-__logf(u) + 1e-20f);
            float le  = (e == i1) ? -__builtin_inff() : acc[e] + gum;
            if (le > best2) { best2 = le; i2 = e; }
        }
        float g1 = g[i1], g2 = g[i2];
        float den = fmaxf(g1 + g2, 1.1920928955078125e-07f);
        float rw1 = g1 / den, rw2 = g2 / den;

        int p1 = atomicAdd(&counts[i1], 1);
        tok_list[i1 * NTOK + p1] = t;  wt_list[i1 * NTOK + p1] = rw1;
        int p2 = atomicAdd(&counts[i2], 1);
        tok_list[i2 * NTOK + p2] = t;  wt_list[i2 * NTOK + p2] = rw2;
    }
}

// ---------------- kernel 4: fused expert FFN (WMMA bf16) ----------------
// grid = E * (NTOK/TM) blocks, 256 threads (8 waves).
__launch_bounds__(256)
__global__ void moe_ffn(const __bf16* __restrict__ xb,    // (N,D)   bf16
                        const __bf16* __restrict__ w1b,   // (E,H,D) bf16
                        const __bf16* __restrict__ w2b,   // (E,D,H) bf16
                        const int*   __restrict__ counts,
                        const int*   __restrict__ tok_list,
                        const float* __restrict__ wt_list,
                        float* __restrict__ out) {
    __shared__ __align__(32) __bf16 xs[TM * XS_STRIDE];        //  66.0 KB
    __shared__ __align__(32) __bf16 hs[TM * HS_STRIDE];        //  16.9 KB
    __shared__ __align__(32) char   w1s[8][2][32 * CSTRIDE];   //  40.0 KB
    __shared__ __align__(32) char   w2s[8][2][128 * CSTRIDE];  // 160.0 KB
    __shared__ int   s_tok[TM];
    __shared__ float s_wt[TM];

    const int tiles_per_e = NTOK / TM;
    const int e    = blockIdx.x / tiles_per_e;
    const int tile = blockIdx.x % tiles_per_e;
    const int cnt  = counts[e];
    const int m0   = tile * TM;
    if (m0 >= cnt) return;
    const int rows = min(TM, cnt - m0);

    const int tid  = threadIdx.x;
    const int lane = tid & 31;
    const int wave = tid >> 5;
    const int half = lane >> 4;
    const int nl   = lane & 15;

    if (tid < TM) {
        if (tid < rows) {
            s_tok[tid] = tok_list[e * NTOK + m0 + tid];
            s_wt[tid]  = wt_list[e * NTOK + m0 + tid];
        } else { s_tok[tid] = -1; s_wt[tid] = 0.0f; }
    }
    __syncthreads();

    // --- async gather of token rows straight into LDS -------------------
    {
        const unsigned lds_base = lds_addr_of(xs);
#pragma unroll
        for (int j = 0; j < 4; ++j) {
            int gidx = tid + 256 * j;
            int r    = gidx >> 5;             // row 0..31
            int c32  = (gidx & 31) << 5;      // element offset (32 bf16 = 64B)
            int t    = s_tok[r]; if (t < 0) t = 0;  // garbage rows discarded
            async_copy_64B(lds_base + (unsigned)((r * XS_STRIDE + c32) * 2),
                           xb + (long)t * DDIM + c32);
        }
        wait_async0();
    }
    __syncthreads();

    const __bf16* w1e = w1b + (long)e * HDIM * DDIM;
    const __bf16* w2e = w2b + (long)e * DDIM * HDIM;

    v8f acc[2][8];
    v8f zf = {};
#pragma unroll
    for (int m = 0; m < 2; ++m)
#pragma unroll
        for (int n = 0; n < 8; ++n) acc[m][n] = zf;

    const int dbase = wave * 128;
    const unsigned w1base[2] = { lds_addr_of(&w1s[wave][0][0]),
                                 lds_addr_of(&w1s[wave][1][0]) };
    const unsigned w2base[2] = { lds_addr_of(&w2s[wave][0][0]),
                                 lds_addr_of(&w2s[wave][1][0]) };

    for (int hc = 0; hc < HDIM / HCHUNK; ++hc) {
        // ---------- GEMM1: h[:,hbase..hbase+32) = x @ W1[e].T ----------
        const int hbase = hc * HCHUNK + wave * 32;
        const __bf16* w1row = w1e + (long)hbase * DDIM;

        if (hc + 1 < HDIM / HCHUNK) {   // L2 prefetch of next chunk's rows
            __builtin_prefetch(w1row + (long)HCHUNK * DDIM + nl * DDIM, 0, 0);
        }

        v8f c1[2][2];
#pragma unroll
        for (int m = 0; m < 2; ++m)
#pragma unroll
            for (int n = 0; n < 2; ++n) c1[m][n] = zf;

        stage_w1(w1base[0], w1row, lane);              // prefill K=0 slice
#pragma unroll
        for (int s = 0; s < DDIM / 32; ++s) {          // 32 steps, unrolled
            const int kk = s * 32;
            if (s + 1 < DDIM / 32) {
                stage_w1(w1base[(s + 1) & 1], w1row + kk + 32, lane);
                wait_async4();                         // batch s complete
            } else {
                wait_async0();                         // drain final batch
            }
            const char* buf = (const char*)&w1s[wave][s & 1][0];
            v16bf a0 = lds_frag_a(xs, XS_STRIDE, 0,  kk, lane);
            v16bf a1 = lds_frag_a(xs, XS_STRIDE, 16, kk, lane);
            v16bf b0 = lds_frag_bs(buf, nl,      half);
            v16bf b1 = lds_frag_bs(buf, 16 + nl, half);
            c1[0][0] = wmma_bf16(a0, b0, c1[0][0]);
            c1[1][0] = wmma_bf16(a1, b0, c1[1][0]);
            c1[0][1] = wmma_bf16(a0, b1, c1[0][1]);
            c1[1][1] = wmma_bf16(a1, b1, c1[1][1]);
        }

        // exact GELU -> bf16 -> shared h-chunk
#pragma unroll
        for (int m = 0; m < 2; ++m)
#pragma unroll
            for (int n = 0; n < 2; ++n)
#pragma unroll
                for (int v = 0; v < 8; ++v) {
                    int r = m * 16 + half * 8 + v;
                    hs[r * HS_STRIDE + wave * 32 + n * 16 + nl] =
                        (__bf16)gelu_exact(c1[m][n][v]);
                }
        __syncthreads();

        // ---------- GEMM2: acc += h_chunk @ W2[e][:,chunk].T ----------
        const int kglob = hc * HCHUNK;
        const __bf16* w2row = w2e + (long)dbase * HDIM + kglob;

        stage_w2(w2base[0], w2row, lane);              // prefill K=0 slice
#pragma unroll
        for (int s = 0; s < HCHUNK / 32; ++s) {        // 8 steps, unrolled
            const int kk = s * 32;
            if (s + 1 < HCHUNK / 32) {
                stage_w2(w2base[(s + 1) & 1], w2row + kk + 32, lane);
                wait_async16();                        // batch s complete
            } else {
                wait_async0();                         // drain final batch
            }
            const char* buf = (const char*)&w2s[wave][s & 1][0];
            v16bf a0 = lds_frag_a(hs, HS_STRIDE, 0,  kk, lane);
            v16bf a1 = lds_frag_a(hs, HS_STRIDE, 16, kk, lane);
            // one-deep register pipeline over n: issue next B ds_loads
            // above the current wmma pair -> partial s_wait_dscnt
            v16bf bc = lds_frag_bs(buf, nl, half);
#pragma unroll
            for (int n = 0; n < 8; ++n) {
                v16bf bn;
                if (n < 7) bn = lds_frag_bs(buf, (n + 1) * 16 + nl, half);
                acc[0][n] = wmma_bf16(a0, bc, acc[0][n]);
                acc[1][n] = wmma_bf16(a1, bc, acc[1][n]);
                if (n < 7) bc = bn;
            }
        }
        __syncthreads();   // hs overwritten next chunk
    }

    // ---------- epilogue: scale by combine weight, atomic add ----------
#pragma unroll
    for (int m = 0; m < 2; ++m)
#pragma unroll
        for (int v = 0; v < 8; ++v) {
            int r = m * 16 + half * 8 + v;
            if (r < rows) {
                float wt = s_wt[r];
                float* orow = out + (long)s_tok[r] * DDIM + dbase;
#pragma unroll
                for (int n = 0; n < 8; ++n)
                    atomicAdd(&orow[n * 16 + nl], wt * acc[m][n][v]);
            }
        }
}

// ---------------- host launcher ----------------
extern "C" void kernel_launch(void* const* d_in, const int* in_sizes, int n_in,
                              void* d_out, int out_size, void* d_ws, size_t ws_size,
                              hipStream_t stream) {
    const float* x  = (const float*)d_in[0];
    const float* gw = (const float*)d_in[1];
    const float* w1 = (const float*)d_in[2];
    const float* w2 = (const float*)d_in[3];
    const float* gu = (const float*)d_in[4];
    float* out = (float*)d_out;

    const long WN = (long)EEXP * HDIM * DDIM;
    const long XN = (long)NTOK * DDIM;

    char* ws = (char*)d_ws;
    __bf16* w1b     = (__bf16*)(ws);                       //  64 MB
    __bf16* w2b     = (__bf16*)(ws + WN * 2);              //  64 MB
    int*    counts  = (int*)   (ws + WN * 4);
    int*    tok_lst = (int*)   (ws + WN * 4 + 64);         // 256 KB
    float*  wt_lst  = (float*) (ws + WN * 4 + 64 + (long)EEXP * NTOK * 4);
    __bf16* xb      = (__bf16*)(ws + WN * 4 + 64 + (long)EEXP * NTOK * 8); // 16 MB

    moe_init<<<2048, 256, 0, stream>>>(out, counts);
    moe_cvt_bf16<<<4096, 256, 0, stream>>>(w1, w1b, WN);
    moe_cvt_bf16<<<4096, 256, 0, stream>>>(w2, w2b, WN);
    moe_cvt_bf16<<<1024, 256, 0, stream>>>(x, xb, XN);
    moe_route<<<NTOK / 8, 256, 0, stream>>>(x, gw, gu, counts, tok_lst, wt_lst);
    moe_ffn<<<EEXP * (NTOK / TM), 256, 0, stream>>>(xb, w1b, w2b, counts,
                                                    tok_lst, wt_lst, out);
}
